// DeltaNetTemporal_6055903888124
// MI455X (gfx1250) — matrix operand order repaired and checked by reading, not compiled, not run on hardware
//
#include <hip/hip_runtime.h>
#include <math.h>

// ---- problem dims (fixed by reference) ----
static constexpr int Bb  = 16;
static constexpr int Nn  = 4096;
static constexpr int Cc  = 256;
static constexpr int Hh  = 4;
static constexpr int Dd  = 64;
static constexpr int BN  = Bb * Nn;          // 65536 rows
static constexpr int BNC = BN * Cc;          // 16777216 elements
static constexpr int SSZ = Bb * Hh * Dd * Dd; // 262144 state elements

typedef __attribute__((ext_vector_type(16))) __bf16 v16bf;
typedef __attribute__((ext_vector_type(8)))  __bf16 v8bf;
typedef __attribute__((ext_vector_type(8)))  float  v8f;

// ---- bf16 <-> f32 (round-to-nearest-even) ----
__device__ __forceinline__ __bf16 f2bf(float f) {
    unsigned u = __float_as_uint(f);
    unsigned r = (u + 0x7fffu + ((u >> 16) & 1u)) >> 16;
    unsigned short s = (unsigned short)r;
    return __builtin_bit_cast(__bf16, s);
}
__device__ __forceinline__ float bf2f(__bf16 b) {
    unsigned short s = __builtin_bit_cast(unsigned short, b);
    return __uint_as_float(((unsigned)s) << 16);
}

// ---- WMMA fragment loaders (contiguous k order k = 16*(lane/16) + s, same for A and B) ----
// A: 16x32 bf16, row-major src, lda elements. lane<16: row m0+lane, K {k0..k0+15}
//                                             lane>=16: same rows, K {k0+16..k0+31}
__device__ __forceinline__ v16bf load_a_frag(const __bf16* __restrict__ A, int lda, int m0, int k0) {
    int lane = threadIdx.x & 31;
    const __bf16* p = A + (m0 + (lane & 15)) * lda + k0 + ((lane >> 4) << 4);
    v8bf lo = *reinterpret_cast<const v8bf*>(p);
    v8bf hi = *reinterpret_cast<const v8bf*>(p + 8);
    v16bf f;
#pragma unroll
    for (int i = 0; i < 8; ++i) { f[i] = lo[i]; f[i + 8] = hi[i]; }
    return f;
}
// B: 32x16 where B[k][n] = W[n][k] (W row-major N x K). lane = col n0+(lane&15),
// 16 contiguous K starting at k0 + 16*(lane>>4)  -> one 32B load.
__device__ __forceinline__ v16bf load_b_frag(const __bf16* __restrict__ W, int ldb, int n0, int k0) {
    int lane = threadIdx.x & 31;
    const __bf16* p = W + (n0 + (lane & 15)) * ldb + k0 + ((lane >> 4) << 4);
    return *reinterpret_cast<const v16bf*>(p);
}
__device__ __forceinline__ v8f wmma_bf16(v16bf a, v16bf b, v8f c) {
    return __builtin_amdgcn_wmma_f32_16x16x32_bf16(false, a, false, b, (short)0, c, false, false);
}

// ---- elementwise f32 -> bf16 ----
__global__ void k_f32_to_bf16(const float* __restrict__ src, __bf16* __restrict__ dst, int n) {
    int i = blockIdx.x * blockDim.x + threadIdx.x;
    if (i < n) dst[i] = f2bf(src[i]);
}

__global__ void k_zero(float* __restrict__ p, int n) {
    int i = blockIdx.x * blockDim.x + threadIdx.x;
    if (i < n) p[i] = 0.f;
}

// ---- fused q/k/v projection GEMM: out[m,n] = sum_k x[m,k] * W[n,k] ----
// One wave -> 16 rows x 64 cols, 3 weight matrices (A frag reused 12x per k-step).
__global__ void __launch_bounds__(128) k_gemm_qkv(
    const __bf16* __restrict__ xb,
    const __bf16* __restrict__ wq, const __bf16* __restrict__ wk, const __bf16* __restrict__ wv,
    __bf16* __restrict__ qb, float* __restrict__ kraw, float* __restrict__ vbuf)
{
    int wave = blockIdx.x * (blockDim.x >> 5) + (threadIdx.x >> 5);
    int lane = threadIdx.x & 31;
    int n0 = (wave & 3) * 64;
    int m0 = (wave >> 2) * 16;

    v8f acc[3][4] = {};
#pragma unroll
    for (int k0 = 0; k0 < Cc; k0 += 32) {
        v16bf a = load_a_frag(xb, Cc, m0, k0);
        const __bf16* Ws[3] = { wq, wk, wv };
#pragma unroll
        for (int w = 0; w < 3; ++w) {
#pragma unroll
            for (int nt = 0; nt < 4; ++nt) {
                v16bf bb = load_b_frag(Ws[w], Cc, n0 + nt * 16, k0);
                acc[w][nt] = wmma_bf16(a, bb, acc[w][nt]);
            }
        }
    }
    int colb  = lane & 15;
    int rbase = m0 + ((lane >> 4) << 3);
#pragma unroll
    for (int nt = 0; nt < 4; ++nt) {
        int col = n0 + nt * 16 + colb;
#pragma unroll
        for (int r = 0; r < 8; ++r) {
            int idx = (rbase + r) * Cc + col;
            qb[idx]   = f2bf(acc[0][nt][r]);
            kraw[idx] = acc[1][nt][r];
            vbuf[idx] = acc[2][nt][r];
        }
    }
}

// ---- k feature map: elu(k)+1, L2-normalize over D=64 per (b,n,h); write bf16 ----
__global__ void __launch_bounds__(128) k_kfeat(const float* __restrict__ kraw, __bf16* __restrict__ kn) {
    int wave = blockIdx.x * (blockDim.x >> 5) + (threadIdx.x >> 5);
    int lane = threadIdx.x & 31;
    int h  = wave & 3;
    int bn = wave >> 2;                    // b*N + n
    int base = bn * Cc + h * Dd;
    float x0 = kraw[base + lane];
    float x1 = kraw[base + 32 + lane];
    float e0 = x0 > 0.f ? x0 + 1.f : __expf(x0);   // elu(x)+1
    float e1 = x1 > 0.f ? x1 + 1.f : __expf(x1);
    float ss = e0 * e0 + e1 * e1;
#pragma unroll
    for (int o = 16; o; o >>= 1) ss += __shfl_xor(ss, o, 32);
    float inv = 1.f / (sqrtf(ss) + 1e-6f);
    kn[base + lane]      = f2bf(e0 * inv);
    kn[base + 32 + lane] = f2bf(e1 * inv);
}

// ---- beta = sigmoid(x @ Wb^T): one wave per row, 4 heads ----
__global__ void __launch_bounds__(128) k_beta(const float* __restrict__ x,
                                              const float* __restrict__ Wb,
                                              float* __restrict__ beta) {
    int wave = blockIdx.x * (blockDim.x >> 5) + (threadIdx.x >> 5);
    int lane = threadIdx.x & 31;
    int base = wave * Cc;
    float a0 = 0.f, a1 = 0.f, a2 = 0.f, a3 = 0.f;
#pragma unroll
    for (int t = 0; t < 8; ++t) {
        int c = lane + t * 32;
        float xv = x[base + c];
        a0 += xv * Wb[c];
        a1 += xv * Wb[Cc + c];
        a2 += xv * Wb[2 * Cc + c];
        a3 += xv * Wb[3 * Cc + c];
    }
#pragma unroll
    for (int o = 16; o; o >>= 1) {
        a0 += __shfl_xor(a0, o, 32); a1 += __shfl_xor(a1, o, 32);
        a2 += __shfl_xor(a2, o, 32); a3 += __shfl_xor(a3, o, 32);
    }
    if (lane == 0) {
        beta[wave * 4 + 0] = 1.f / (1.f + __expf(-a0));
        beta[wave * 4 + 1] = 1.f / (1.f + __expf(-a1));
        beta[wave * 4 + 2] = 1.f / (1.f + __expf(-a2));
        beta[wave * 4 + 3] = 1.f / (1.f + __expf(-a3));
    }
}

// ---- w_key sums: wkey_sum[b,c] = sum_n kn[b,n,c] ----
__global__ void __launch_bounds__(256) k_wkey(const __bf16* __restrict__ kn, float* __restrict__ wkey) {
    int b = blockIdx.x >> 4;
    int n0 = (blockIdx.x & 15) * 256;
    int c = threadIdx.x;
    float s = 0.f;
    for (int n = 0; n < 256; ++n) s += bf2f(kn[(b * Nn + n0 + n) * Cc + c]);
    atomicAdd(&wkey[b * Cc + c], s);
}

// ---- state read: y = q @ Sd^T, pred = k @ Sd^T (fused), plus w_err partial sums ----
__global__ void __launch_bounds__(128) k_state(
    const __bf16* __restrict__ qb, const __bf16* __restrict__ kn,
    const float* __restrict__ S, const float* __restrict__ v, const float* __restrict__ beta,
    float* __restrict__ y, float* __restrict__ werr)
{
    int wave = blockIdx.x * (blockDim.x >> 5) + (threadIdx.x >> 5);
    int lane = threadIdx.x & 31;
    int i0 = (wave & 3) * 16;
    int m0 = ((wave >> 2) & 255) * 16;
    int h  = (wave >> 10) & 3;
    int b  = wave >> 12;

    const float* Sb = S + (b * Hh + h) * Dd * Dd;
    const __bf16* qbase = qb + b * Nn * Cc + h * Dd;
    const __bf16* kbase = kn + b * Nn * Cc + h * Dd;

    v8f accy = {}, accp = {};
#pragma unroll
    for (int k0 = 0; k0 < Dd; k0 += 32) {
        // B fragment = Sd^T: B[j,i] = 0.95*S[i,j]; per lane fixed i, 16 contiguous j
        v16bf bf;
        {
            int i  = i0 + (lane & 15);
            int j0 = k0 + ((lane >> 4) << 4);
            const float* p = Sb + i * Dd + j0;
#pragma unroll
            for (int e = 0; e < 16; ++e) bf[e] = f2bf(0.95f * p[e]);
        }
        v16bf aq = load_a_frag(qbase, Cc, m0, k0);
        v16bf ak = load_a_frag(kbase, Cc, m0, k0);
        accy = wmma_bf16(aq, bf, accy);
        accp = wmma_bf16(ak, bf, accp);
    }
    int icol  = i0 + (lane & 15);
    int rbase = m0 + ((lane >> 4) << 3);
    float es = 0.f;
#pragma unroll
    for (int r = 0; r < 8; ++r) {
        int m = rbase + r;
        int idx = (b * Nn + m) * Cc + h * Dd + icol;
        y[idx] = accy[r];                              // merged-head layout directly
        es += beta[(b * Nn + m) * 4 + h] * (v[idx] - accp[r]);
    }
    es += __shfl_xor(es, 16, 32);                      // lanes l and l+16 share column icol
    if (lane < 16) atomicAdd(&werr[b * Cc + h * Dd + i0 + lane], es);
}

// ---- S_new = clip(0.95*S + mean(w_err) (x) mean(w_key), +-10) ----
__global__ void k_snew(const float* __restrict__ S, const float* __restrict__ werr,
                       const float* __restrict__ wkey, float* __restrict__ outS) {
    int id = blockIdx.x * blockDim.x + threadIdx.x;
    if (id >= SSZ) return;
    int j = id & 63;
    int i = (id >> 6) & 63;
    int h = (id >> 12) & 3;
    int b = id >> 14;
    float sd = 0.95f * S[id];
    float we = werr[b * Cc + h * Dd + i] * (1.f / (float)Nn);
    float wk = wkey[b * Cc + h * Dd + j] * (1.f / (float)Nn);
    float sn = sd + we * wk;
    outS[id] = fminf(10.f, fmaxf(-10.f, sn));
}

// ---- RMSNorm over C + g_rms, write bf16 for final GEMM ----
__global__ void __launch_bounds__(128) k_rms(const float* __restrict__ y, const float* __restrict__ g,
                                             __bf16* __restrict__ yb) {
    int wave = blockIdx.x * (blockDim.x >> 5) + (threadIdx.x >> 5);
    int lane = threadIdx.x & 31;
    int base = wave * Cc;
    float vals[8];
    float ss = 0.f;
#pragma unroll
    for (int t = 0; t < 8; ++t) { vals[t] = y[base + lane + t * 32]; ss += vals[t] * vals[t]; }
#pragma unroll
    for (int o = 16; o; o >>= 1) ss += __shfl_xor(ss, o, 32);
    float r = rsqrtf(ss * (1.f / (float)Cc) + 1e-6f);
#pragma unroll
    for (int t = 0; t < 8; ++t) {
        int c = lane + t * 32;
        yb[base + c] = f2bf(vals[t] * r * g[c]);
    }
}

// ---- output projection: out[m,n] = sum_k yb[m,k] * Wo[n,k] ----
__global__ void __launch_bounds__(128) k_gemm_out(const __bf16* __restrict__ yb,
                                                  const __bf16* __restrict__ wo,
                                                  float* __restrict__ out) {
    int wave = blockIdx.x * (blockDim.x >> 5) + (threadIdx.x >> 5);
    int lane = threadIdx.x & 31;
    int n0 = (wave & 3) * 64;
    int m0 = (wave >> 2) * 16;
    v8f acc[4] = {};
#pragma unroll
    for (int k0 = 0; k0 < Cc; k0 += 32) {
        v16bf a = load_a_frag(yb, Cc, m0, k0);
#pragma unroll
        for (int nt = 0; nt < 4; ++nt) {
            v16bf bb = load_b_frag(wo, Cc, n0 + nt * 16, k0);
            acc[nt] = wmma_bf16(a, bb, acc[nt]);
        }
    }
    int colb  = lane & 15;
    int rbase = m0 + ((lane >> 4) << 3);
#pragma unroll
    for (int nt = 0; nt < 4; ++nt) {
        int col = n0 + nt * 16 + colb;
#pragma unroll
        for (int r = 0; r < 8; ++r) out[(rbase + r) * Cc + col] = acc[nt][r];
    }
}

extern "C" void kernel_launch(void* const* d_in, const int* in_sizes, int n_in,
                              void* d_out, int out_size, void* d_ws, size_t ws_size,
                              hipStream_t stream) {
    (void)in_sizes; (void)n_in; (void)out_size; (void)ws_size;
    const float* x  = (const float*)d_in[0];
    const float* S  = (const float*)d_in[1];
    const float* Wq = (const float*)d_in[2];
    const float* Wk = (const float*)d_in[3];
    const float* Wv = (const float*)d_in[4];
    const float* Wb = (const float*)d_in[5];
    const float* Wo = (const float*)d_in[6];
    const float* g  = (const float*)d_in[7];
    float* out = (float*)d_out;

    char* ws = (char*)d_ws;
    size_t off = 0;
    __bf16* xb   = (__bf16*)(ws + off); off += (size_t)BNC * 2;   // aliased as yb later
    __bf16* qb   = (__bf16*)(ws + off); off += (size_t)BNC * 2;
    __bf16* kn   = (__bf16*)(ws + off); off += (size_t)BNC * 2;
    float*  kraw = (float*) (ws + off); off += (size_t)BNC * 4;   // aliased as y later
    float*  vbuf = (float*) (ws + off); off += (size_t)BNC * 4;
    float*  beta = (float*) (ws + off); off += (size_t)BN * Hh * 4;
    __bf16* wqb  = (__bf16*)(ws + off); off += (size_t)Cc * Cc * 2;
    __bf16* wkb  = (__bf16*)(ws + off); off += (size_t)Cc * Cc * 2;
    __bf16* wvb  = (__bf16*)(ws + off); off += (size_t)Cc * Cc * 2;
    __bf16* wob  = (__bf16*)(ws + off); off += (size_t)Cc * Cc * 2;
    float*  werr = (float*) (ws + off); off += (size_t)Bb * Cc * 4;
    float*  wkey = (float*) (ws + off); off += (size_t)Bb * Cc * 4;
    float*  y    = kraw;   // kraw dead after k_kfeat
    __bf16* yb   = xb;     // xb dead after k_gemm_qkv

    // 1. precision conversion (one pass over x; weights tiny, L2-resident in bf16)
    k_f32_to_bf16<<<(BNC + 255) / 256, 256, 0, stream>>>(x, xb, BNC);
    k_f32_to_bf16<<<(Cc * Cc + 255) / 256, 256, 0, stream>>>(Wq, wqb, Cc * Cc);
    k_f32_to_bf16<<<(Cc * Cc + 255) / 256, 256, 0, stream>>>(Wk, wkb, Cc * Cc);
    k_f32_to_bf16<<<(Cc * Cc + 255) / 256, 256, 0, stream>>>(Wv, wvb, Cc * Cc);
    k_f32_to_bf16<<<(Cc * Cc + 255) / 256, 256, 0, stream>>>(Wo, wob, Cc * Cc);
    k_zero<<<(2 * Bb * Cc + 255) / 256, 256, 0, stream>>>(werr, 2 * Bb * Cc); // werr+wkey contiguous

    // 2. fused q/k/v projections (WMMA)
    k_gemm_qkv<<<(BN / 16) * 4 / 4, 128, 0, stream>>>(xb, wqb, wkb, wvb, qb, kraw, vbuf);
    // 3. key feature map + L2 norm
    k_kfeat<<<BN * Hh / 4, 128, 0, stream>>>(kraw, kn);
    // 4. per-head gates
    k_beta<<<BN / 4, 128, 0, stream>>>(x, Wb, beta);
    // 5. w_key sums
    k_wkey<<<Bb * 16, 256, 0, stream>>>(kn, wkey);
    // 6. state read (y, pred) + w_err sums (WMMA)
    k_state<<<(Bb * Hh * 256 * 4) / 4, 128, 0, stream>>>(qb, kn, S, vbuf, beta, y, werr);
    // 7. state update -> d_out tail
    k_snew<<<(SSZ + 255) / 256, 256, 0, stream>>>(S, werr, wkey, out + BNC);
    // 8. RMSNorm + bf16
    k_rms<<<BN / 4, 128, 0, stream>>>(y, g, yb);
    // 9. output projection (WMMA) -> d_out head
    k_gemm_out<<<(BN / 16) * 4 / 4, 128, 0, stream>>>(yb, wob, out);
}
